// CausalSelfAttention_10247791968738
// MI455X (gfx1250) — matrix operand order repaired and checked by reading, not compiled, not run on hardware
//
#include <hip/hip_runtime.h>

// ---------------- CDNA5 / gfx1250 WMMA types ----------------
typedef __attribute__((ext_vector_type(16))) __bf16 v16bf;
typedef __attribute__((ext_vector_type(8)))  float  v8f;
typedef __attribute__((ext_vector_type(4)))  unsigned int u32x4;
typedef __attribute__((ext_vector_type(8)))  unsigned int u32x8;
typedef unsigned short u16;

union Frag16 {            // one 16x32 bf16 A/B fragment = 8 VGPRs
    v16bf v;
    uint4 u[2];
};

#define B_    2
#define T_    2048
#define C_    2048
#define H_    16
#define KVH_  4
#define D_    128
#define GRP_  (H_ / KVH_)     // 4
#define EPS_  1.1920929e-07f
#define SCALE_ 0.08838834764831845f   // 1/sqrt(128)

static __device__ inline u16 f2bf(float f) {
    unsigned int u = __builtin_bit_cast(unsigned int, f);
    unsigned int r = u + 0x7FFFu + ((u >> 16) & 1u);   // round-to-nearest-even
    return (u16)(r >> 16);
}

static __device__ inline v8f vzero8() {
    v8f z = {0.f, 0.f, 0.f, 0.f, 0.f, 0.f, 0.f, 0.f};
    return z;
}

// ---------------- f32 -> bf16 elementwise cast (4 elems/thread) ----------------
__global__ __launch_bounds__(256) void cast_f32_bf16(const float* __restrict__ s,
                                                     u16* __restrict__ d, int n) {
    int i = (blockIdx.x * 256 + threadIdx.x) * 4;
    if (i >= n) return;
    float4 f = *(const float4*)(s + i);
    union { u16 h[4]; uint2 v; } o;
    o.h[0] = f2bf(f.x); o.h[1] = f2bf(f.y); o.h[2] = f2bf(f.z); o.h[3] = f2bf(f.w);
    *(uint2*)(d + i) = o.v;
}

// ---------------- GEMM: C(MxN, f32) = A(MxK, bf16) * B(NxK, bf16)^T ----------------
// Tile 128x128x32, 256 threads = 8 waves; wave (wm= wid&3 -> 32 rows, wn= wid>>2 -> 64 cols)
__global__ __launch_bounds__(256) void gemm_bt_bf16(const u16* __restrict__ A,
                                                    const u16* __restrict__ Bw,
                                                    float* __restrict__ Co,
                                                    int M, int N, int K) {
    __shared__ __align__(16) u16 sA[128 * 32];
    __shared__ __align__(16) u16 sB[128 * 32];

    const int tid  = threadIdx.x;
    const int wid  = tid >> 5;
    const int lane = tid & 31;
    const int m16  = lane & 15;
    const int g    = lane >> 4;
    const int wm   = wid & 3;
    const int wn   = wid >> 2;
    const int tileM = blockIdx.y * 128;
    const int tileN = blockIdx.x * 128;

    // per-thread global-load coordinates: 16 bf16 (two uint4) per matrix per k-step
    const int ldRow = tid >> 1;            // 0..127
    const int ldCol = (tid & 1) * 16;      // 0 or 16
    const u16* aG = A  + (size_t)(tileM + ldRow) * K + ldCol;
    const u16* bG = Bw + (size_t)(tileN + ldRow) * K + ldCol;

    v8f acc[2][4];
#pragma unroll
    for (int mt = 0; mt < 2; ++mt)
#pragma unroll
        for (int nt = 0; nt < 4; ++nt) acc[mt][nt] = vzero8();

    const int ksteps = K >> 5;
    for (int ks = 0; ks < ksteps; ++ks) {
        __syncthreads();
        // stage 128x32 bf16 tiles of A and B into LDS
        *(uint4*)(sA + ldRow * 32 + ldCol)     = *(const uint4*)(aG);
        *(uint4*)(sA + ldRow * 32 + ldCol + 8) = *(const uint4*)(aG + 8);
        *(uint4*)(sB + ldRow * 32 + ldCol)     = *(const uint4*)(bG);
        *(uint4*)(sB + ldRow * 32 + ldCol + 8) = *(const uint4*)(bG + 8);
        __builtin_prefetch(aG + 32, 0, 1);     // global_prefetch_b8 of next k-slab
        __builtin_prefetch(bG + 32, 0, 1);
        aG += 32; bG += 32;
        __syncthreads();

        Frag16 af[2];
#pragma unroll
        for (int mt = 0; mt < 2; ++mt) {
            const u16* p = sA + (wm * 32 + mt * 16 + m16) * 32;
            af[mt].u[0] = *(const uint4*)(p + g * 8);        // K = g*8 .. g*8+7
            af[mt].u[1] = *(const uint4*)(p + 16 + g * 8);   // K = 16+g*8 ..
        }
#pragma unroll
        for (int nt = 0; nt < 4; ++nt) {
            Frag16 bf_;
            const u16* p = sB + (wn * 64 + nt * 16 + m16) * 32;
            bf_.u[0] = *(const uint4*)(p + g * 8);           // K = g*8 .. g*8+7
            bf_.u[1] = *(const uint4*)(p + 16 + g * 8);      // K = 16+g*8 ..
#pragma unroll
            for (int mt = 0; mt < 2; ++mt) {
                acc[mt][nt] = __builtin_amdgcn_wmma_f32_16x16x32_bf16(
                    false, af[mt].v, false, bf_.v, (short)0, acc[mt][nt], false, false);
            }
        }
    }

    // epilogue: D layout -> row = r + 8*g, col = lane&15
#pragma unroll
    for (int mt = 0; mt < 2; ++mt)
#pragma unroll
        for (int nt = 0; nt < 4; ++nt) {
            const int col = tileN + wn * 64 + nt * 16 + m16;
#pragma unroll
            for (int r = 0; r < 8; ++r) {
                const int row = tileM + wm * 32 + mt * 16 + r + 8 * g;
                Co[(size_t)row * N + col] = acc[mt][nt][r];
            }
        }
}

// ---------------- fused RoPE + RMSNorm, f32 [B,T,NH,128] -> bf16 [B,NH,T,128] ----------------
__global__ __launch_bounds__(256) void rope_rms_bf16(const float* __restrict__ src,
                                                     const float* __restrict__ cp,
                                                     const float* __restrict__ sp,
                                                     u16* __restrict__ dst, int NH) {
    const int wid  = threadIdx.x >> 5;
    const int lane = threadIdx.x & 31;
    const int row  = blockIdx.x * 8 + wid;          // < B*T*NH
    const int h = row % NH;
    const int t = (row / NH) % T_;
    const int b = row / (NH * T_);
    const float* r = src + (size_t)row * 128;

    float x1a = r[lane],      x1b = r[lane + 32];
    float x2a = r[lane + 64], x2b = r[lane + 96];
    float ca = cp[t * 64 + lane],      sa = sp[t * 64 + lane];
    float cb = cp[t * 64 + lane + 32], sb = sp[t * 64 + lane + 32];
    float o1a = x1a * ca + x2a * sa, o2a = x2a * ca - x1a * sa;
    float o1b = x1b * cb + x2b * sb, o2b = x2b * cb - x1b * sb;

    float ss = o1a * o1a + o1b * o1b + o2a * o2a + o2b * o2b;
#pragma unroll
    for (int off = 16; off > 0; off >>= 1) ss += __shfl_xor(ss, off, 32);
    const float rn = rsqrtf(ss * (1.f / 128.f) + EPS_);

    u16* o = dst + ((size_t)(b * NH + h) * T_ + t) * 128;
    o[lane]      = f2bf(o1a * rn);
    o[lane + 32] = f2bf(o1b * rn);
    o[lane + 64] = f2bf(o2a * rn);
    o[lane + 96] = f2bf(o2b * rn);
}

// ---------------- V: f32 [B,T,NH,128] -> bf16 [B,NH,T,128] ----------------
__global__ __launch_bounds__(256) void perm_cast_bf16(const float* __restrict__ src,
                                                      u16* __restrict__ dst, int NH) {
    const int wid  = threadIdx.x >> 5;
    const int lane = threadIdx.x & 31;
    const int row  = blockIdx.x * 8 + wid;
    const int h = row % NH;
    const int t = (row / NH) % T_;
    const int b = row / (NH * T_);
    const float* r = src + (size_t)row * 128;
    u16* o = dst + ((size_t)(b * NH + h) * T_ + t) * 128;
#pragma unroll
    for (int i = 0; i < 4; ++i) o[lane + i * 32] = f2bf(r[lane + i * 32]);
}

// ---------------- causal flash attention, bf16 WMMA, f32 online softmax ----------------
// q [B,H,T,128], k/v [B,KVH,T,128]  ->  y bf16 [B,T,H*128]
// 64 q-rows per block; 8 waves = 4 row-groups x 2 key-halves of the 64-key tile.
// K tile is staged global->LDS by the Tensor Data Mover (tensor_load_to_lds, TENSORcnt);
// V tile is staged transposed by the threads (TDM cannot transpose).
__global__ __launch_bounds__(256) void flash_attn_bf16(const u16* __restrict__ qb,
                                                       const u16* __restrict__ kb,
                                                       const u16* __restrict__ vb,
                                                       u16* __restrict__ yb) {
    __shared__ __align__(16) u16  sKV[64 * 128 + 128 * 64]; // sK (64x128) then sVT (128x64); 32 KB
    __shared__ __align__(16) u16  sP[8 * 16 * 32];          // per-wave P tiles; 8 KB
    __shared__ float sM[4][16][2];
    __shared__ float sL[4][16][2];

    u16* sK  = sKV;
    u16* sVT = sKV + 64 * 128;
    float* sO = (float*)sKV;                                // aliased: used only after the loop

    const int tid  = threadIdx.x;
    const int wid  = tid >> 5;
    const int lane = tid & 31;
    const int m16  = lane & 15;
    const int g    = lane >> 4;
    const int rg   = wid & 3;   // row group: 16 q rows
    const int kh   = wid >> 2;  // key half: 32 of 64 keys

    const int bh  = blockIdx.y;
    const int b   = bh / H_;
    const int h   = bh % H_;
    const int kvh = h / GRP_;
    const int qt0 = blockIdx.x * 64;

    const u16* Qp = qb + ((size_t)(b * H_ + h)     * T_) * 128;
    const u16* Kp = kb + ((size_t)(b * KVH_ + kvh) * T_) * 128;
    const u16* Vp = vb + ((size_t)(b * KVH_ + kvh) * T_) * 128;

    // Q fragments for this wave's 16 rows, whole D=128 (4 k-steps), kept in VGPRs
    Frag16 qf[4];
    {
        const u16* qr = Qp + (size_t)(qt0 + rg * 16 + m16) * 128;
#pragma unroll
        for (int ks = 0; ks < 4; ++ks) {
            qf[ks].u[0] = *(const uint4*)(qr + ks * 32 + g * 8);
            qf[ks].u[1] = *(const uint4*)(qr + ks * 32 + 16 + g * 8);
        }
    }

    // Tensor DMA descriptor pieces that do not change across iterations.
    // D# group0: [1:0]=count=1 | [63:32]=lds_addr | [120:64]=global_addr | [127:126]=type(2)
    // D# group1: [17:16]=data_size(1 -> 2B) | tensor_dim0=128 | tensor_dim1=64 |
    //            tile_dim0=128, tile_dim1=64 | tensor_dim0_stride=128
    const unsigned int sk_lds = (unsigned int)(size_t)(&sK[0]); // low 32b of generic = LDS byte addr

    v8f   Oacc[8];
#pragma unroll
    for (int dt = 0; dt < 8; ++dt) Oacc[dt] = vzero8();
    float mrun[8], lrun[8];
#pragma unroll
    for (int r = 0; r < 8; ++r) { mrun[r] = -1e30f; lrun[r] = 0.f; }

    const int jmax = qt0 / 64;
    for (int j = 0; j <= jmax; ++j) {
        const int kbase = j * 64;

        // ---- K tile: one TDM transfer issued by wave 0 (EXEC-independent, wave-level) ----
        if (wid == 0) {
            unsigned long long ga = (unsigned long long)(size_t)(Kp + (size_t)kbase * 128);
            u32x4 g0;
            g0[0] = 1u;                                          // count=1 (valid user D#)
            g0[1] = sk_lds;                                      // lds_addr
            g0[2] = (unsigned int)(ga & 0xFFFFFFFFu);            // global_addr[31:0]
            g0[3] = (unsigned int)((ga >> 32) & 0x01FFFFFFu)     // global_addr[56:32]
                    | (2u << 30);                                // type = 2 ("image")
            u32x8 g1;
            g1[0] = (1u << 16);          // workgroup_mask=0, data_size=1 (2 bytes)
            g1[1] = (128u << 16);        // atomic_barrier_addr=0; tensor_dim0[15:0]=128
            g1[2] = (64u << 16);         // tensor_dim0[31:16]=0; tensor_dim1[15:0]=64
            g1[3] = (128u << 16);        // tensor_dim1[31:16]=0; tile_dim0=128
            g1[4] = 64u;                 // tile_dim1=64; tile_dim2=0
            g1[5] = 128u;                // tensor_dim0_stride[31:0]=128
            g1[6] = 0u;                  // stride0[47:32]=0; stride1[15:0]=0 (unused, 2D)
            g1[7] = 0u;                  // stride1[47:16]=0
            asm volatile("tensor_load_to_lds %0, %1" :: "s"(g0), "s"(g1) : "memory");
#if __has_builtin(__builtin_amdgcn_s_wait_tensorcnt)
            __builtin_amdgcn_s_wait_tensorcnt(0);
#else
            asm volatile("s_wait_tensorcnt 0x0" ::: "memory");
#endif
        }

        // ---- V tile transposed into LDS by the threads ----
#pragma unroll
        for (int it = 0; it < 32; ++it) {
            int e = tid + it * 256;                 // 8192 elems
            int kr = e >> 7, dd = e & 127;
            sVT[dd * 64 + kr] = Vp[(size_t)(kbase + kr) * 128 + dd];
        }
        __syncthreads();

        // S = Q * K^T (scaled, causal-masked), 16x32 per wave as 2 n-tiles
        float st[2][8];
#pragma unroll
        for (int nt = 0; nt < 2; ++nt) {
            v8f s = vzero8();
#pragma unroll
            for (int ks = 0; ks < 4; ++ks) {
                Frag16 bf_;
                const u16* kr = sK + (kh * 32 + nt * 16 + m16) * 128 + ks * 32 + g * 8;
                bf_.u[0] = *(const uint4*)(kr);
                bf_.u[1] = *(const uint4*)(kr + 16);
                s = __builtin_amdgcn_wmma_f32_16x16x32_bf16(
                        false, qf[ks].v, false, bf_.v, (short)0, s, false, false);
            }
            const int kcol = kbase + kh * 32 + nt * 16 + m16;
#pragma unroll
            for (int r = 0; r < 8; ++r) {
                const int qrow = qt0 + rg * 16 + r + 8 * g;
                float v = s[r] * SCALE_;
                if (kcol > qrow) v = -1e30f;
                st[nt][r] = v;
            }
        }

        // per-row tile max across this wave's 32 keys
        float tmax[8];
#pragma unroll
        for (int r = 0; r < 8; ++r) tmax[r] = fmaxf(st[0][r], st[1][r]);
#pragma unroll
        for (int off = 1; off < 16; off <<= 1)
#pragma unroll
            for (int r = 0; r < 8; ++r) tmax[r] = fmaxf(tmax[r], __shfl_xor(tmax[r], off, 32));
        if (m16 == 0)
#pragma unroll
            for (int r = 0; r < 8; ++r) sM[rg][r + 8 * g][kh] = tmax[r];
        __syncthreads();

        float mnew[8], alpha[8];
#pragma unroll
        for (int r = 0; r < 8; ++r) {
            float mj = fmaxf(sM[rg][r + 8 * g][0], sM[rg][r + 8 * g][1]);
            mnew[r]  = fmaxf(mrun[r], mj);
            alpha[r] = __expf(mrun[r] - mnew[r]);
            mrun[r]  = mnew[r];
        }

        // P = exp(S - m_new); row partial sums; P -> bf16 LDS for A-fragment reload
        float rs[8];
#pragma unroll
        for (int nt = 0; nt < 2; ++nt)
#pragma unroll
            for (int r = 0; r < 8; ++r) st[nt][r] = __expf(st[nt][r] - mnew[r]);
#pragma unroll
        for (int r = 0; r < 8; ++r) rs[r] = st[0][r] + st[1][r];
#pragma unroll
        for (int off = 1; off < 16; off <<= 1)
#pragma unroll
            for (int r = 0; r < 8; ++r) rs[r] += __shfl_xor(rs[r], off, 32);
        if (m16 == 0)
#pragma unroll
            for (int r = 0; r < 8; ++r) sL[rg][r + 8 * g][kh] = rs[r];
#pragma unroll
        for (int nt = 0; nt < 2; ++nt)
#pragma unroll
            for (int r = 0; r < 8; ++r)
                sP[wid * 512 + (r + 8 * g) * 32 + nt * 16 + m16] = f2bf(st[nt][r]);
        __syncthreads();

#pragma unroll
        for (int r = 0; r < 8; ++r)
            lrun[r] = alpha[r] * lrun[r] + sL[rg][r + 8 * g][0] + sL[rg][r + 8 * g][1];
#pragma unroll
        for (int dt = 0; dt < 8; ++dt)
#pragma unroll
            for (int r = 0; r < 8; ++r) Oacc[dt][r] = Oacc[dt][r] * alpha[r];

        // O += P(16x32) * V(32x128): A-frag from sP, B-frags from transposed V
        Frag16 pf;
        {
            const u16* pr = sP + wid * 512 + m16 * 32;
            pf.u[0] = *(const uint4*)(pr + g * 8);
            pf.u[1] = *(const uint4*)(pr + 16 + g * 8);
        }
#pragma unroll
        for (int dt = 0; dt < 8; ++dt) {
            Frag16 vf;
            const u16* vr = sVT + (dt * 16 + m16) * 64 + kh * 32 + g * 8;
            vf.u[0] = *(const uint4*)(vr);
            vf.u[1] = *(const uint4*)(vr + 16);
            Oacc[dt] = __builtin_amdgcn_wmma_f32_16x16x32_bf16(
                           false, pf.v, false, vf.v, (short)0, Oacc[dt], false, false);
        }
        __syncthreads();
    }

    // combine the two key-half partial O's, normalize, write bf16 y [B,T,H*128]
    if (kh == 1) {
#pragma unroll
        for (int dt = 0; dt < 8; ++dt)
#pragma unroll
            for (int r = 0; r < 8; ++r)
                sO[(rg * 16 + r + 8 * g) * 128 + dt * 16 + m16] = Oacc[dt][r];
    }
    __syncthreads();
    if (kh == 0) {
#pragma unroll
        for (int dt = 0; dt < 8; ++dt)
#pragma unroll
            for (int r = 0; r < 8; ++r) {
                const int rloc = r + 8 * g;
                float tot = Oacc[dt][r] + sO[(rg * 16 + rloc) * 128 + dt * 16 + m16];
                float o   = tot / lrun[r];
                const int t = qt0 + rg * 16 + rloc;
                yb[((size_t)(b * T_) + t) * 2048 + h * 128 + dt * 16 + m16] = f2bf(o);
            }
    }
}

// ---------------- host launcher ----------------
extern "C" void kernel_launch(void* const* d_in, const int* in_sizes, int n_in,
                              void* d_out, int out_size, void* d_ws, size_t ws_size,
                              hipStream_t stream) {
    (void)in_sizes; (void)n_in; (void)out_size; (void)ws_size;
    const float* x    = (const float*)d_in[0];
    const float* cosp = (const float*)d_in[1];
    const float* sinp = (const float*)d_in[2];
    const float* Wq   = (const float*)d_in[3];
    const float* Wk   = (const float*)d_in[4];
    const float* Wv   = (const float*)d_in[5];
    const float* Wo   = (const float*)d_in[6];
    float* out = (float*)d_out;

    char* ws = (char*)d_ws;
    const size_t MB = 1ull << 20;
    u16*   xb  = (u16*)(ws + 0);          //  16 MB  bf16 x      [4096,2048]
    u16*   wqb = (u16*)(ws + 16 * MB);    //   8 MB  bf16 Wq     [2048,2048]
    u16*   wkb = (u16*)(ws + 24 * MB);    //   2 MB  bf16 Wk     [512,2048]
    u16*   wvb = (u16*)(ws + 26 * MB);    //   2 MB  bf16 Wv
    u16*   wob = (u16*)(ws + 28 * MB);    //   8 MB  bf16 Wo
    float* qf  = (float*)(ws + 36 * MB);  //  32 MB  f32 q       [B,T,H,128]
    float* kf  = (float*)(ws + 68 * MB);  //   8 MB  f32 k       [B,T,KVH,128]
    float* vf  = (float*)(ws + 76 * MB);  //   8 MB  f32 v
    u16*   qhb = (u16*)(ws + 84 * MB);    //  16 MB  bf16 q head-major [B,H,T,128]
    u16*   khb = (u16*)(ws + 100 * MB);   //   4 MB  bf16 k head-major
    u16*   vhb = (u16*)(ws + 104 * MB);   //   4 MB  bf16 v head-major
    u16*   yb  = xb;                      //  alias: x is dead after the QKV GEMMs

    // 1) cast inputs/weights to bf16 (all sizes divisible by 1024)
    cast_f32_bf16<<<(B_ * T_ * C_) / 1024, 256, 0, stream>>>(x,  xb,  B_ * T_ * C_);
    cast_f32_bf16<<<(H_ * D_ * C_) / 1024, 256, 0, stream>>>(Wq, wqb, H_ * D_ * C_);
    cast_f32_bf16<<<(KVH_ * D_ * C_) / 1024, 256, 0, stream>>>(Wk, wkb, KVH_ * D_ * C_);
    cast_f32_bf16<<<(KVH_ * D_ * C_) / 1024, 256, 0, stream>>>(Wv, wvb, KVH_ * D_ * C_);
    cast_f32_bf16<<<(C_ * C_) / 1024, 256, 0, stream>>>(Wo, wob, C_ * C_);

    // 2) QKV projections: q = x*Wq^T, k = x*Wk^T, v = x*Wv^T  (bf16 WMMA, f32 accum)
    gemm_bt_bf16<<<dim3(2048 / 128, 4096 / 128), 256, 0, stream>>>(xb, wqb, qf, 4096, 2048, 2048);
    gemm_bt_bf16<<<dim3(512  / 128, 4096 / 128), 256, 0, stream>>>(xb, wkb, kf, 4096, 512,  2048);
    gemm_bt_bf16<<<dim3(512  / 128, 4096 / 128), 256, 0, stream>>>(xb, wvb, vf, 4096, 512,  2048);

    // 3) RoPE + RMSNorm (q,k) and cast+permute (v) into head-major bf16
    rope_rms_bf16<<<(B_ * T_ * H_)   / 8, 256, 0, stream>>>(qf, cosp, sinp, qhb, H_);
    rope_rms_bf16<<<(B_ * T_ * KVH_) / 8, 256, 0, stream>>>(kf, cosp, sinp, khb, KVH_);
    perm_cast_bf16<<<(B_ * T_ * KVH_) / 8, 256, 0, stream>>>(vf, vhb, KVH_);

    // 4) causal flash attention (bf16 WMMA; K tile staged via Tensor Data Mover)
    flash_attn_bf16<<<dim3(T_ / 64, B_ * H_), 256, 0, stream>>>(qhb, khb, vhb, yb);

    // 5) output projection: out = y * Wo^T (f32 result to d_out)
    gemm_bt_bf16<<<dim3(2048 / 128, 4096 / 128), 256, 0, stream>>>(yb, wob, out, 4096, 2048, 2048);
}
